// MultiheadAttentionBlock_51436528337577
// MI455X (gfx1250) — matrix operand order repaired and checked by reading, not compiled, not run on hardware
//
#include <hip/hip_runtime.h>
#include <hip/hip_bf16.h>

// Problem constants (match reference)
#define B_  2
#define S_  2048
#define D_  1024
#define H_  16
#define DK_ 64
#define BS_ (B_ * S_)   // 4096

typedef __attribute__((ext_vector_type(16))) _Float16 v16h;
typedef __attribute__((ext_vector_type(8)))  _Float16 v8h;
typedef __attribute__((ext_vector_type(4)))  _Float16 v4h;
typedef __attribute__((ext_vector_type(2)))  _Float16 v2h;
typedef __attribute__((ext_vector_type(8)))  float    v8f;

// ---------------------------------------------------------------------------
// Fragment loaders for v_wmma_f32_16x16x32_f16 (wave32).
// A (16x32, MxK): lanes 0-15 row M=lane (K=0..15); lanes 16-31 row M=lane-16
//                 (K=16..31).
// B (32x16, KxN): lane n holds column n; elements i = K = kb + i (kb=16 upper half).
// B is always read from a row-major matrix M[N][K] with B[k][n] = M[n][k], so every
// lane reads 16 contiguous f16 (32B): matches x@W.T, Q·K^T (row-major K), and P·V
// with V stored transposed.
// ---------------------------------------------------------------------------

static __device__ __forceinline__ v16h load_a16x32(const _Float16* base, int ld) {
  const int lane = threadIdx.x & 31;
  const int m    = lane & 15;
  const int kb   = (lane >> 4) << 4;           // 0 or 16
  const _Float16* p = base + m * ld + kb;
  v8h lo = *(const v8h*)(p);
  v8h hi = *(const v8h*)(p + 8);
  v16h a;
#pragma unroll
  for (int i = 0; i < 8; ++i) { a[i] = lo[i]; a[8 + i] = hi[i]; }
  return a;
}

static __device__ __forceinline__ v16h load_bT32x16(const _Float16* base, int ld) {
  const int lane = threadIdx.x & 31;
  const int n    = lane & 15;
  const int kb   = (lane >> 4) << 4;           // 0 or 16
  return *(const v16h*)(base + n * ld + kb);
}

static __device__ __forceinline__ v8f wmma_f16(v16h a, v16h b, v8f c) {
  return __builtin_amdgcn_wmma_f32_16x16x32_f16(false, a, false, b, (short)0, c,
                                                false, false);
}

// XOR-swap within 16-lane halves via ds_swizzle (pattern in immediate:
// offset = xor_mask<<10 | or_mask<<5 | and_mask, and_mask=0x1f).
static __device__ __forceinline__ v2h pkmax(v2h a, v2h b) {
  return __builtin_elementwise_max(a, b);
}
#define SWZ_STAGE(pkv, imm)                                                    \
  do {                                                                         \
    _Pragma("unroll") for (int p_ = 0; p_ < 4; ++p_) {                         \
      int s_ = __builtin_amdgcn_ds_swizzle(__builtin_bit_cast(int, pkv[p_]),   \
                                           (imm));                             \
      pkv[p_] = pkmax(pkv[p_], __builtin_bit_cast(v2h, s_));                   \
    }                                                                          \
  } while (0)

// ---------------------------------------------------------------------------
// fp32 -> f16 conversion (vectorized x4)
// ---------------------------------------------------------------------------
__global__ void cvt_f32_to_f16(const float* __restrict__ in,
                               _Float16* __restrict__ out, int n) {
  int i = (blockIdx.x * blockDim.x + threadIdx.x) * 4;
  if (i + 3 < n) {
    float4 v = *(const float4*)(in + i);
    v4h h;
    h[0] = (_Float16)v.x; h[1] = (_Float16)v.y;
    h[2] = (_Float16)v.z; h[3] = (_Float16)v.w;
    *(v4h*)(out + i) = h;
  }
}

// ---------------------------------------------------------------------------
// GEMM: C[M,N] = (X[M,K] * W[N,K]^T + bias) * scale   (M=BS_, K=N=D_)
// One wave -> 16x64 output tile. 4 waves / block.
// MODE 0: store f16 into [B,H,S,DK]      (Q with scale=1/8, K with scale=1)
// MODE 1: store f16 into [B,H,DK,S]      (V transposed for P·V fragments)
// MODE 2: store f32 row-major [BS, D]    (final output projection)
// ---------------------------------------------------------------------------
template <int MODE>
__global__ __launch_bounds__(128, 1)
void gemm16x64(const _Float16* __restrict__ X,
               const _Float16* __restrict__ W,
               const float* __restrict__ bias,
               void* __restrict__ out, float scale) {
  const int wid   = threadIdx.x >> 5;
  const int wglob = blockIdx.x * 4 + wid;      // 4096 waves
  const int rt    = wglob >> 4;                // 256 row tiles
  const int ct    = wglob & 15;                // 16 col tiles (64 wide)
  const int row0  = rt * 16;
  const int n0    = ct * 64;
  const int lane  = threadIdx.x & 31;
  const int nl    = lane & 15;
  const int hi    = lane >> 4;

  v8f acc[4] = {};
  const _Float16* xrow = X + (size_t)row0 * D_;

  for (int k0 = 0; k0 < D_; k0 += 32) {
    v16h a = load_a16x32(xrow + k0, D_);
#pragma unroll
    for (int t = 0; t < 4; ++t) {
      v16h b = load_bT32x16(W + (size_t)(n0 + t * 16) * D_ + k0, D_);
      acc[t] = wmma_f16(a, b, acc[t]);
    }
  }

  const int bidx = row0 / S_;                  // batch (tiles never straddle)
  const int s0   = row0 % S_;
  const int h    = n0 / DK_;                   // head (64-wide tile == 1 head)

#pragma unroll
  for (int t = 0; t < 4; ++t) {
    float bv = bias[n0 + t * 16 + nl];
#pragma unroll
    for (int i = 0; i < 8; ++i) {
      int   r  = i + 8 * hi;                   // local row 0..15
      float vv = (acc[t][i] + bv) * scale;
      if (MODE == 0) {
        _Float16* o = (_Float16*)out;
        int dk = t * 16 + nl;
        o[(((size_t)bidx * H_ + h) * S_ + (s0 + r)) * DK_ + dk] = (_Float16)vv;
      } else if (MODE == 1) {
        _Float16* o = (_Float16*)out;
        int dk = t * 16 + nl;
        o[(((size_t)bidx * H_ + h) * DK_ + dk) * S_ + (s0 + r)] = (_Float16)vv;
      } else {
        float* o = (float*)out;
        o[(size_t)(row0 + r) * D_ + n0 + t * 16 + nl] = vv;
      }
    }
  }
}

// ---------------------------------------------------------------------------
// Flash attention (causal), one 16-row q-tile per wave, k-tiles of 32.
// qh (pre-scaled by 1/sqrt(DK)) / kh: [B,H,S,DK] f16 ; vt: [B,H,DK,S] f16
// xat out: [B,S,D] f16
// - Only the single diagonal tile is masked (peeled predicate).
// - Row max: packed-f16 ds_swizzle XOR tree (16 DS ops/tile).
// - Row sum: extra WMMA against an all-ones B fragment (lacc behaves exactly
//   like an O column with V == 1, including the alpha rescale).
// ---------------------------------------------------------------------------
__global__ __launch_bounds__(128, 1)
void flash_attn(const _Float16* __restrict__ qh,
                const _Float16* __restrict__ kh,
                const _Float16* __restrict__ vt,
                _Float16* __restrict__ xat) {
  __shared__ __align__(32) _Float16 lds_p[4][16 * 32];   // per-wave P staging

  const int wid   = threadIdx.x >> 5;
  const int wglob = blockIdx.x * 4 + wid;      // B*H*(S/16) = 4096 waves
  const int qt    = wglob & 127;               // S/16 = 128 q-tiles
  const int bh    = wglob >> 7;                // b*H + h
  const int b     = bh >> 4;
  const int h     = bh & 15;
  const int q0    = qt * 16;
  const int lane  = threadIdx.x & 31;
  const int nl    = lane & 15;
  const int hi    = lane >> 4;

  const _Float16* qbase = qh + ((size_t)bh * S_ + q0) * DK_;
  const _Float16* kbase = kh + (size_t)bh * S_ * DK_;
  const _Float16* vbase = vt + (size_t)bh * DK_ * S_;

  // Q fragments for DK=64 (two K=32 steps), loaded once (already pre-scaled)
  v16h aq0 = load_a16x32(qbase, DK_);
  v16h aq1 = load_a16x32(qbase + 32, DK_);

  // all-ones B fragment for the row-sum WMMA
  v16h bones;
#pragma unroll
  for (int i = 0; i < 16; ++i) bones[i] = (_Float16)1.0f;

  v8f   o[4] = {};
  v8f   lacc = {};
  float mrow[8];
#pragma unroll
  for (int i = 0; i < 8; ++i) mrow[i] = -1e30f;

  _Float16* lp = &lds_p[wid][0];
  const int jend = (q0 + 15) >> 5;             // the single diagonal (masked) tile

  for (int j = 0; j <= jend; ++j) {
    const int kpos0 = j << 5;
    const bool diag = (j == jend);

    // ---- S = Q K^T (16x32 tile via two 16-col subtiles, 4 WMMAs) ----
    v8f s0 = {}, s1 = {};
    {
      const _Float16* kb0 = kbase + (size_t)kpos0 * DK_;
      v16h b0 = load_bT32x16(kb0, DK_);
      v16h b1 = load_bT32x16(kb0 + 16 * DK_, DK_);
      s0 = wmma_f16(aq0, b0, s0);
      s1 = wmma_f16(aq0, b1, s1);
      b0 = load_bT32x16(kb0 + 32, DK_);
      b1 = load_bT32x16(kb0 + 16 * DK_ + 32, DK_);
      s0 = wmma_f16(aq1, b0, s0);
      s1 = wmma_f16(aq1, b1, s1);
    }

    // ---- causal mask (diagonal tile only) ----
    if (diag) {
#pragma unroll
      for (int i = 0; i < 8; ++i) {
        int qg = q0 + i + 8 * hi;
        if (kpos0 + nl      > qg) s0[i] = -1e30f;
        if (kpos0 + 16 + nl > qg) s1[i] = -1e30f;
      }
    }

    // ---- row max: pack 8 row maxima as f16 pairs, 4 XOR swizzle stages ----
    v2h pk[4];
#pragma unroll
    for (int p = 0; p < 4; ++p) {
      v2h t;
      t[0] = (_Float16)fmaxf(s0[2 * p],     s1[2 * p]);
      t[1] = (_Float16)fmaxf(s0[2 * p + 1], s1[2 * p + 1]);
      pk[p] = t;
    }
    SWZ_STAGE(pk, 0x041F);   // xor 1
    SWZ_STAGE(pk, 0x081F);   // xor 2
    SWZ_STAGE(pk, 0x101F);   // xor 4
    SWZ_STAGE(pk, 0x201F);   // xor 8  (stays within each 16-lane half)

    // ---- online update: exp in place, rescale O and l, stage P to LDS ----
#pragma unroll
    for (int i = 0; i < 8; ++i) {
      float rm   = (float)pk[i >> 1][i & 1];
      float mnew = fmaxf(mrow[i], rm);
      float al   = __expf(mrow[i] - mnew);
      mrow[i]    = mnew;
      float e0   = __expf(s0[i] - mnew);
      float e1   = __expf(s1[i] - mnew);
      o[0][i] *= al; o[1][i] *= al; o[2][i] *= al; o[3][i] *= al;
      lacc[i] *= al;
      int r = i + 8 * hi;
      lp[r * 32 + nl]      = (_Float16)e0;
      lp[r * 32 + 16 + nl] = (_Float16)e1;
    }
    asm volatile("s_wait_dscnt 0" ::: "memory");   // wave-private; DS in-order

    // ---- reload P as A fragment; O += P·V ; l += P·1 (5 WMMAs) ----
    v16h ap = load_a16x32(lp, 32);
    lacc = wmma_f16(ap, bones, lacc);
#pragma unroll
    for (int t = 0; t < 4; ++t) {
      v16h bv = load_bT32x16(vbase + (size_t)(t * 16) * S_ + kpos0, S_);
      o[t] = wmma_f16(ap, bv, o[t]);
    }
  }

  // ---- normalize and store to [B,S,D] f16 ----
  float inv[8];
#pragma unroll
  for (int i = 0; i < 8; ++i) inv[i] = 1.0f / lacc[i];
#pragma unroll
  for (int t = 0; t < 4; ++t)
#pragma unroll
    for (int i = 0; i < 8; ++i) {
      int r    = i + 8 * hi;
      int srow = q0 + r;
      int dcol = h * DK_ + t * 16 + nl;
      xat[((size_t)b * S_ + srow) * D_ + dcol] = (_Float16)(o[t][i] * inv[i]);
    }
}

// ---------------------------------------------------------------------------
extern "C" void kernel_launch(void* const* d_in, const int* in_sizes, int n_in,
                              void* d_out, int out_size, void* d_ws, size_t ws_size,
                              hipStream_t stream) {
  const float* q  = (const float*)d_in[0];
  const float* k  = (const float*)d_in[1];
  const float* v  = (const float*)d_in[2];
  // d_in[3] = causal mask (implicit in flash kernel)
  const float* Wq = (const float*)d_in[4];
  const float* bq = (const float*)d_in[5];
  const float* Wk = (const float*)d_in[6];
  const float* bk = (const float*)d_in[7];
  const float* Wv = (const float*)d_in[8];
  const float* bv = (const float*)d_in[9];
  const float* Wo = (const float*)d_in[10];
  const float* bo = (const float*)d_in[11];

  const size_t NE = (size_t)BS_ * D_;   // 4,194,304
  const size_t NW = (size_t)D_ * D_;    // 1,048,576

  _Float16* ws  = (_Float16*)d_ws;      // needs 32M f16 = 64 MB
  _Float16* qc  = ws;
  _Float16* kc  = qc + NE;
  _Float16* vc  = kc + NE;
  _Float16* wq  = vc + NE;
  _Float16* wk  = wq + NW;
  _Float16* wv  = wk + NW;
  _Float16* wo  = wv + NW;
  _Float16* qhp = wo + NW;              // [B,H,S,DK]  (pre-scaled by 1/8)
  _Float16* khp = qhp + NE;             // [B,H,S,DK]
  _Float16* vtp = khp + NE;             // [B,H,DK,S]
  _Float16* xat = vtp + NE;             // [B,S,D]

  const int CT = 256;
  cvt_f32_to_f16<<<(int)(NE / (CT * 4)), CT, 0, stream>>>(q, qc, (int)NE);
  cvt_f32_to_f16<<<(int)(NE / (CT * 4)), CT, 0, stream>>>(k, kc, (int)NE);
  cvt_f32_to_f16<<<(int)(NE / (CT * 4)), CT, 0, stream>>>(v, vc, (int)NE);
  cvt_f32_to_f16<<<(int)(NW / (CT * 4)), CT, 0, stream>>>(Wq, wq, (int)NW);
  cvt_f32_to_f16<<<(int)(NW / (CT * 4)), CT, 0, stream>>>(Wk, wk, (int)NW);
  cvt_f32_to_f16<<<(int)(NW / (CT * 4)), CT, 0, stream>>>(Wv, wv, (int)NW);
  cvt_f32_to_f16<<<(int)(NW / (CT * 4)), CT, 0, stream>>>(Wo, wo, (int)NW);

  // 4096 waves each: (BS/16 row tiles) x (D/64 col tiles); 4 waves/block
  gemm16x64<0><<<1024, 128, 0, stream>>>(qc, wq, bq, qhp, 0.125f); // 1/sqrt(64)
  gemm16x64<0><<<1024, 128, 0, stream>>>(kc, wk, bk, khp, 1.0f);
  gemm16x64<1><<<1024, 128, 0, stream>>>(vc, wv, bv, vtp, 1.0f);

  // B*H*(S/16) = 4096 q-tile waves; 4 waves/block
  flash_attn<<<1024, 128, 0, stream>>>(qhp, khp, vtp, xat);

  gemm16x64<2><<<1024, 128, 0, stream>>>(xat, wo, bo, d_out, 1.0f);
}
